// MultiHeadAttention_44178033606761
// MI455X (gfx1250) — matrix operand order repaired and checked by reading, not compiled, not run on hardware
//
#include <hip/hip_runtime.h>
#include <hip/hip_bf16.h>

#define BSZ    4
#define SEQ    2048
#define DMODEL 512
#define HEADS  8
#define HD     64
#define NTOK   (BSZ*SEQ)         // 8192
#define NEGV   (-1000000.0f)
#define INV_SCALE 0.125f         // 1/sqrt(64)

typedef __attribute__((ext_vector_type(16))) __bf16 bf16x16;
typedef __attribute__((ext_vector_type(8)))  __bf16 bf16x8;
typedef __attribute__((ext_vector_type(4)))  __bf16 bf16x4;
typedef __attribute__((ext_vector_type(8)))  float  f32x8;
typedef __attribute__((ext_vector_type(4)))  float  f32x4;
typedef __attribute__((ext_vector_type(4)))  int    i32x4;

union BFrag  { bf16x16 v; bf16x8 h[2]; };
union BPack8 { bf16x8  v; __bf16 e[8]; };

// ---------------- CDNA5 async global->LDS staging ----------------
#if defined(__has_builtin)
#  if __has_builtin(__builtin_amdgcn_global_load_async_to_lds_b128)
#    define HAVE_ASYNC_LDS 1
#  endif
#endif
#ifndef HAVE_ASYNC_LDS
#  define HAVE_ASYNC_LDS 0
#endif

static __device__ inline void async_copy16(const __bf16* g, __bf16* l) {
#if HAVE_ASYNC_LDS
  __builtin_amdgcn_global_load_async_to_lds_b128(
      (__attribute__((address_space(1))) i32x4*)g,
      (__attribute__((address_space(3))) i32x4*)l, 0, 0);
#else
  *(bf16x8*)l = *(const bf16x8*)g;   // sync fallback
#endif
}
static __device__ inline void async_join() {
#if HAVE_ASYNC_LDS
  asm volatile("s_wait_asynccnt 0x0" ::: "memory");
#endif
}

static __device__ inline f32x8 wmma_bf16(bf16x16 a, bf16x16 b, f32x8 c) {
  return __builtin_amdgcn_wmma_f32_16x16x32_bf16(false, a, false, b, (short)0, c,
                                                 false, false);
}

// A fragment: 16x32 (MxK) bf16, contiguous k = 16*hi + s per lane (B uses the same order).
static __device__ inline bf16x16 load_a_frag(const __bf16* base, int ld, int lane) {
  int laneM = lane & 15, hi = lane >> 4;
  const __bf16* p = base + (size_t)laneM * ld + hi * 16;
  BFrag f;
  f.h[0] = *(const bf16x8*)(p);
  f.h[1] = *(const bf16x8*)(p + 8);
  return f.v;
}

// B fragment from an LDS tile stored as CS[n][k] row-contiguous over k.
static __device__ inline bf16x16 lds_b_frag(const __bf16* p /* = &tile[n=laneN][hi*16] */) {
  BFrag f;
  f.h[0] = *(const bf16x8*)(p);
  f.h[1] = *(const bf16x8*)(p + 8);
  return f.v;
}

// ---------------- fp32 -> bf16 conversion, 4-wide ----------------
__global__ __launch_bounds__(256) void cvt_f32_bf16_x4(const float* __restrict__ src,
                                                       __bf16* __restrict__ dst, int n4) {
  int i = blockIdx.x * 256 + threadIdx.x;
  if (i < n4) {
    f32x4 v = ((const f32x4*)src)[i];
    bf16x4 o;
#pragma unroll
    for (int j = 0; j < 4; ++j) o[j] = (__bf16)v[j];
    ((bf16x4*)dst)[i] = o;
  }
}

// ---------------- QKV projection: C(8192x512) = Xb @ W^T ----------------
// grid = 3 * 32 row-tiles(256) * 8 col-tiles(64). 8 waves * 32 rows each.
// B tile (64 cols x 32 K, 4KB) staged in LDS via async copy, double-buffered.
__global__ __launch_bounds__(256) void qkv_gemm(
    const __bf16* __restrict__ xb,
    const __bf16* __restrict__ wqb, const __bf16* __restrict__ wkb,
    const __bf16* __restrict__ wvb,
    __bf16* __restrict__ qb, __bf16* __restrict__ kb, __bf16* __restrict__ vtb) {
  __shared__ __align__(16) __bf16 bstage[2][64][32];
  int tid  = threadIdx.x;
  int lane = tid & 31, wave = tid >> 5;
  int w     = blockIdx.x / 256;        // 0=Q 1=K 2=V
  int rem   = blockIdx.x % 256;
  int rtile = rem >> 3;                // 0..31
  int ctile = rem & 7;                 // 0..7
  int r0 = rtile * 256 + wave * 32;
  int j0 = ctile * 64;
  const __bf16* W = (w == 0) ? wqb : (w == 1) ? wkb : wvb;

  int srow = tid >> 2;                 // 0..63  (staging row = output col j)
  int scol = (tid & 3) * 8;            // staging chunk

  f32x8 zero = {};
  f32x8 acc[2][4];
#pragma unroll
  for (int hlf = 0; hlf < 2; ++hlf)
#pragma unroll
    for (int t = 0; t < 4; ++t) acc[hlf][t] = zero;

  // prologue: stage K-step 0
  async_copy16(W + (size_t)(j0 + srow) * DMODEL + scol, &bstage[0][srow][scol]);

  int laneN = lane & 15, hi = lane >> 4;
  for (int i = 0; i < DMODEL / 32; ++i) {
    int kk = i * 32;
    async_join();
    __syncthreads();
    if (i + 1 < DMODEL / 32)
      async_copy16(W + (size_t)(j0 + srow) * DMODEL + kk + 32 + scol,
                   &bstage[(i + 1) & 1][srow][scol]);

    bf16x16 a0 = load_a_frag(xb + (size_t)r0 * DMODEL + kk,        DMODEL, lane);
    bf16x16 a1 = load_a_frag(xb + (size_t)(r0 + 16) * DMODEL + kk, DMODEL, lane);
    int buf = i & 1;
#pragma unroll
    for (int t = 0; t < 4; ++t) {
      bf16x16 b = lds_b_frag(&bstage[buf][t * 16 + laneN][hi * 16]);
      acc[0][t] = wmma_bf16(a0, b, acc[0][t]);
      acc[1][t] = wmma_bf16(a1, b, acc[1][t]);
    }
  }

  // ---- stores ----
#pragma unroll
  for (int hlf = 0; hlf < 2; ++hlf) {
    int rbase = r0 + hlf * 16;
#pragma unroll
    for (int t = 0; t < 4; ++t) {
      int j = j0 + t * 16 + laneN;
      int h = j >> 6, d = j & 63;
      if (w < 2) {
        __bf16* dst = (w == 0) ? qb : kb;         // (b,h,n,d)
#pragma unroll
        for (int v = 0; v < 8; ++v) {
          int i = rbase + hi * 8 + v;
          int b = i >> 11, n = i & 2047;
          dst[(((size_t)(b * HEADS + h)) * SEQ + n) * HD + d] = (__bf16)acc[hlf][t][v];
        }
      } else {                                    // V transposed: (b,h,d,n)
        int i0 = rbase + hi * 8;
        int b = i0 >> 11, n0 = i0 & 2047;
        BPack8 pk;
#pragma unroll
        for (int v = 0; v < 8; ++v) pk.e[v] = (__bf16)acc[hlf][t][v];
        *(bf16x8*)(vtb + (((size_t)(b * HEADS + h)) * HD + d) * SEQ + n0) = pk.v;
      }
    }
  }
}

// ---------------- streaming-softmax attention ----------------
// Block = 8 waves sharing one (b,h); wave owns 16 queries. K/V tiles for 32
// keys staged in LDS (async, double-buffered) and shared by all 8 waves.
__global__ __launch_bounds__(256) void attn_kernel(
    const __bf16* __restrict__ qb, const __bf16* __restrict__ kb,
    const __bf16* __restrict__ vtb, const int* __restrict__ mask,
    __bf16* __restrict__ attb) {
  __shared__ __align__(16) __bf16 kstage[2][32][64];  // [key][d]
  __shared__ __align__(16) __bf16 vstage[2][64][32];  // [d][key]
  __shared__ __align__(16) __bf16 pstage[8][16 * 32];
  int tid = threadIdx.x;
  int lane = tid & 31, wave = tid >> 5;
  int id = blockIdx.x * 8 + wave;      // 0..4095 ; same bh for whole block
  int bh = id >> 7;
  int qblock = id & 127;
  int b = bh >> 3;
  int q0 = qblock * 16;

  const __bf16* Qbase  = qb  + ((size_t)bh * SEQ + q0) * HD;
  const __bf16* Kbase  = kb  + (size_t)bh * SEQ * HD;
  const __bf16* Vtbase = vtb + (size_t)bh * HD * SEQ;
  const int*    mrow   = mask + b * SEQ;

  int laneM = lane & 15, hi = lane >> 4;
  int krow = tid >> 3, kcol = (tid & 7) * 8;   // K staging: 32x64
  int vrow = tid >> 2, vcol = (tid & 3) * 8;   // V staging: 64x32

  bf16x16 aq0 = load_a_frag(Qbase,      HD, lane);
  bf16x16 aq1 = load_a_frag(Qbase + 32, HD, lane);

  bf16x16 ones;
#pragma unroll
  for (int e = 0; e < 16; ++e) ones[e] = (__bf16)1.0f;

  f32x8 zero = {};
  f32x8 o[4], L = zero;
#pragma unroll
  for (int t = 0; t < 4; ++t) o[t] = zero;
  float m[8];
#pragma unroll
  for (int v = 0; v < 8; ++v) m[v] = -1e30f;

  __bf16* ps = pstage[wave];

  // prologue staging for key-block 0
  async_copy16(Kbase + (size_t)krow * HD + kcol, &kstage[0][krow][kcol]);
  async_copy16(Vtbase + (size_t)vrow * SEQ + vcol, &vstage[0][vrow][vcol]);

  for (int i = 0; i < SEQ / 32; ++i) {
    int kk = i * 32;
    async_join();
    __syncthreads();
    if (i + 1 < SEQ / 32) {
      async_copy16(Kbase + (size_t)(kk + 32 + krow) * HD + kcol,
                   &kstage[(i + 1) & 1][krow][kcol]);
      async_copy16(Vtbase + (size_t)vrow * SEQ + kk + 32 + vcol,
                   &vstage[(i + 1) & 1][vrow][vcol]);
    }
    int buf = i & 1;

    // ---- scores ----
    f32x8 s0 = zero, s1 = zero;
    s0 = wmma_bf16(aq0, lds_b_frag(&kstage[buf][laneM][hi * 16]),           s0);
    s0 = wmma_bf16(aq1, lds_b_frag(&kstage[buf][laneM][32 + hi * 16]),      s0);
    s1 = wmma_bf16(aq0, lds_b_frag(&kstage[buf][16 + laneM][hi * 16]),      s1);
    s1 = wmma_bf16(aq1, lds_b_frag(&kstage[buf][16 + laneM][32 + hi * 16]), s1);

    bool keep0 = mrow[kk + laneM] != 0;
    bool keep1 = mrow[kk + 16 + laneM] != 0;

    // ---- online softmax (row max via shuffles; row sum via ones-WMMA) ----
#pragma unroll
    for (int v = 0; v < 8; ++v) {
      float x0 = keep0 ? s0[v] * INV_SCALE : NEGV;
      float x1 = keep1 ? s1[v] * INV_SCALE : NEGV;
      float rmax = fmaxf(x0, x1);
#pragma unroll
      for (int off = 1; off < 16; off <<= 1)
        rmax = fmaxf(rmax, __shfl_xor(rmax, off, 32));
      float mn   = fmaxf(m[v], rmax);
      float corr = __expf(m[v] - mn);
      m[v] = mn;
      s0[v] = __expf(x0 - mn);
      s1[v] = __expf(x1 - mn);
      L[v] *= corr;
      o[0][v] *= corr; o[1][v] *= corr; o[2][v] *= corr; o[3][v] *= corr;
    }

    // ---- P (C layout) -> A layout via per-wave LDS transpose ----
#pragma unroll
    for (int v = 0; v < 8; ++v) {
      int row = hi * 8 + v;
      ps[row * 32 + laneM]      = (__bf16)s0[v];
      ps[row * 32 + 16 + laneM] = (__bf16)s1[v];
    }
    asm volatile("s_wait_dscnt 0" ::: "memory");
    BFrag pf;
    const __bf16* pp = ps + laneM * 32 + hi * 16;
    pf.h[0] = *(const bf16x8*)(pp);
    pf.h[1] = *(const bf16x8*)(pp + 8);

    // ---- row-sum accumulator and out += P @ V ----
    L = wmma_bf16(pf.v, ones, L);
#pragma unroll
    for (int t = 0; t < 4; ++t) {
      bf16x16 bv = lds_b_frag(&vstage[buf][t * 16 + laneM][hi * 16]);
      o[t] = wmma_bf16(pf.v, bv, o[t]);
    }
  }

  // ---- normalize + store (b,h,n,d) bf16 ----
  float rl[8];
#pragma unroll
  for (int v = 0; v < 8; ++v) rl[v] = 1.0f / L[v];
#pragma unroll
  for (int t = 0; t < 4; ++t)
#pragma unroll
    for (int v = 0; v < 8; ++v)
      attb[((size_t)bh * SEQ + q0 + hi * 8 + v) * HD + t * 16 + laneM] =
          (__bf16)(o[t][v] * rl[v]);
}

// ---------------- output projection: out(8192x512) = Att @ Wo^T (fp32) ----
__global__ __launch_bounds__(256) void out_gemm(
    const __bf16* __restrict__ attb, const __bf16* __restrict__ wob,
    float* __restrict__ out) {
  __shared__ __align__(16) __bf16 bstage[2][64][32];
  int tid = threadIdx.x;
  int lane = tid & 31, wave = tid >> 5;
  int rtile = blockIdx.x >> 3, ctile = blockIdx.x & 7;   // grid 32*8
  int r0 = rtile * 256 + wave * 32;
  int j0 = ctile * 64;
  int laneM = lane & 15, hi = lane >> 4;
  int srow = tid >> 2, scol = (tid & 3) * 8;

  f32x8 zero = {};
  f32x8 acc[2][4];
#pragma unroll
  for (int hlf = 0; hlf < 2; ++hlf)
#pragma unroll
    for (int t = 0; t < 4; ++t) acc[hlf][t] = zero;

  // A rows (token indices) for the two 16-row halves
  const __bf16* arow[2];
#pragma unroll
  for (int hlf = 0; hlf < 2; ++hlf) {
    int i = r0 + hlf * 16 + laneM;
    int b = i >> 11, n = i & 2047;
    arow[hlf] = attb + (size_t)b * HEADS * SEQ * HD + (size_t)n * HD;
  }

  async_copy16(wob + (size_t)(j0 + srow) * DMODEL + scol, &bstage[0][srow][scol]);

  for (int i = 0; i < DMODEL / 32; ++i) {
    int kk = i * 32;
    async_join();
    __syncthreads();
    if (i + 1 < DMODEL / 32)
      async_copy16(wob + (size_t)(j0 + srow) * DMODEL + kk + 32 + scol,
                   &bstage[(i + 1) & 1][srow][scol]);

    int c0 = kk + hi * 16;                // chunks never cross a 64-wide head
    int c1 = kk + hi * 16 + 8;
    BFrag a[2];
#pragma unroll
    for (int hlf = 0; hlf < 2; ++hlf) {
      a[hlf].h[0] = *(const bf16x8*)(arow[hlf] + (size_t)(c0 >> 6) * SEQ * HD + (c0 & 63));
      a[hlf].h[1] = *(const bf16x8*)(arow[hlf] + (size_t)(c1 >> 6) * SEQ * HD + (c1 & 63));
    }
    int buf = i & 1;
#pragma unroll
    for (int t = 0; t < 4; ++t) {
      bf16x16 b = lds_b_frag(&bstage[buf][t * 16 + laneM][hi * 16]);
      acc[0][t] = wmma_bf16(a[0].v, b, acc[0][t]);
      acc[1][t] = wmma_bf16(a[1].v, b, acc[1][t]);
    }
  }

#pragma unroll
  for (int hlf = 0; hlf < 2; ++hlf)
#pragma unroll
    for (int t = 0; t < 4; ++t)
#pragma unroll
      for (int v = 0; v < 8; ++v)
        out[(size_t)(r0 + hlf * 16 + hi * 8 + v) * DMODEL + j0 + t * 16 + laneM] =
            acc[hlf][t][v];
}

extern "C" void kernel_launch(void* const* d_in, const int* in_sizes, int n_in,
                              void* d_out, int out_size, void* d_ws, size_t ws_size,
                              hipStream_t stream) {
  const float* x  = (const float*)d_in[0];
  const float* Wq = (const float*)d_in[1];
  const float* Wk = (const float*)d_in[2];
  const float* Wv = (const float*)d_in[3];
  const float* Wo = (const float*)d_in[4];
  const int* mask = (const int*)d_in[5];
  float* out = (float*)d_out;

  const size_t XE = (size_t)NTOK * DMODEL;   // 4,194,304
  const size_t WE = (size_t)DMODEL * DMODEL; //   262,144
  __bf16* xb   = (__bf16*)d_ws;
  __bf16* wqb  = xb  + XE;
  __bf16* wkb  = wqb + WE;
  __bf16* wvb  = wkb + WE;
  __bf16* wob  = wvb + WE;
  __bf16* qb   = wob + WE;
  __bf16* kb   = qb  + XE;
  __bf16* vtb  = kb  + XE;
  __bf16* attb = vtb + XE;

  cvt_f32_bf16_x4<<<(int)(XE / 4 / 256), 256, 0, stream>>>(x,  xb,  (int)(XE / 4));
  cvt_f32_bf16_x4<<<(int)(WE / 4 / 256), 256, 0, stream>>>(Wq, wqb, (int)(WE / 4));
  cvt_f32_bf16_x4<<<(int)(WE / 4 / 256), 256, 0, stream>>>(Wk, wkb, (int)(WE / 4));
  cvt_f32_bf16_x4<<<(int)(WE / 4 / 256), 256, 0, stream>>>(Wv, wvb, (int)(WE / 4));
  cvt_f32_bf16_x4<<<(int)(WE / 4 / 256), 256, 0, stream>>>(Wo, wob, (int)(WE / 4));

  qkv_gemm<<<768, 256, 0, stream>>>(xb, wqb, wkb, wvb, qb, kb, vtb);
  attn_kernel<<<512, 256, 0, stream>>>(qb, kb, vtb, mask, attb);
  out_gemm<<<256, 256, 0, stream>>>(attb, wob, out);
}
